// ResidualAttentionBlock_27865747817243
// MI455X (gfx1250) — matrix operand order repaired and checked by reading, not compiled, not run on hardware
//
#include <hip/hip_runtime.h>
#include <hip/hip_bf16.h>

typedef __attribute__((ext_vector_type(16))) _Float16 v16h;
typedef __attribute__((ext_vector_type(8)))  _Float16 v8h;
typedef __attribute__((ext_vector_type(8)))  float    v8f;

// ---- gfx1250 async LDS-load path (guarded: falls back to ds_store staging) --
#if defined(__has_builtin)
#  if __has_builtin(__builtin_amdgcn_global_load_async_to_lds_b128)
#    define GEMM_ASYNC_LDS 1
#  endif
#endif
#ifndef GEMM_ASYNC_LDS
#  define GEMM_ASYNC_LDS 0
#endif

// Builtin signature (from hipcc diagnostic): param0 = AS1 (global, printed
// '__device__') pointer to GNU vector int[4]; param1 = AS3 (LDS) pointer.
typedef int v4i_gnu __attribute__((vector_size(16)));
typedef __attribute__((address_space(1))) v4i_gnu AS1v4i;
typedef __attribute__((address_space(3))) v4i_gnu AS3v4i;

namespace {

constexpr int LSEQ   = 256;
constexpr int BATCH  = 64;
constexpr int DMODEL = 1024;
constexpr int NHEAD  = 16;
constexpr int HDIM   = 64;       // DMODEL / NHEAD
constexpr int NEXP   = 8;
constexpr int RADP   = 64;
constexpr int NADP   = NEXP * RADP;   // 512
constexpr int DFF    = 4 * DMODEL;    // 4096
constexpr int MTOK   = LSEQ * BATCH;  // 16384
constexpr float SCALE_ADAPTER = 0.1f;

constexpr int SW_STRIDE = 40;              // halves per LDS W-tile row (80B, bank-spread)
constexpr int SW_ELEMS  = 64 * SW_STRIDE;  // halves per buffer

// ---------------------------------------------------------------- WMMA core

__device__ __forceinline__ v8f wmma32(v16h a, v16h b, v8f c) {
  // D = A(16x32 f16) * B(32x16 f16) + C(16x16 f32)
  return __builtin_amdgcn_wmma_f32_16x16x32_f16(
      /*neg_a=*/false, a, /*neg_b=*/false, b,
      /*c_mod=*/(short)0, c, /*reuse_a=*/false, /*reuse_b=*/false);
}

// A-operand fragment, 16x32 f16, row-major, leading dim ld.
// lane: m = l&15 ; halves i -> K = k0 + (l>>4)*16 + i  (contiguous 16)
__device__ __forceinline__ v16h load_fragA(const _Float16* __restrict__ base,
                                           int row0, int ld, int k0) {
  const int lane = threadIdx.x & 31;
  const _Float16* p = base + (size_t)(row0 + (lane & 15)) * ld + k0 + ((lane >> 4) << 4);
  v8h lo = *(const v8h*)p;
  v8h hi = *(const v8h*)(p + 8);
  v16h f;
#pragma unroll
  for (int i = 0; i < 8; ++i) { f[i] = lo[i]; f[i + 8] = hi[i]; }
  return f;
}

// B-operand fragment (32x16) from rows of W (N x K row-major):
// lane: n = l&15 ; halves i -> K = k0 + (l>>4)*16 + i  (contiguous 16)
__device__ __forceinline__ v16h load_fragB(const _Float16* __restrict__ base,
                                           int row0, int ld, int k0) {
  const int lane = threadIdx.x & 31;
  const _Float16* p = base + (size_t)(row0 + (lane & 15)) * ld + k0 + ((lane >> 4) << 4);
  v8h lo = *(const v8h*)p;
  v8h hi = *(const v8h*)(p + 8);
  v16h f;
#pragma unroll
  for (int i = 0; i < 8; ++i) { f[i] = lo[i]; f[i + 8] = hi[i]; }
  return f;
}

// B-operand fragment from an LDS-staged 64x32 tile (row stride SW_STRIDE halves)
__device__ __forceinline__ v16h load_fragB_lds(const _Float16* smem, int row0) {
  const int lane = threadIdx.x & 31;
  const _Float16* p = smem + (row0 + (lane & 15)) * SW_STRIDE + ((lane >> 4) << 4);
  v8h lo = *(const v8h*)p;
  v8h hi = *(const v8h*)(p + 8);
  v16h f;
#pragma unroll
  for (int i = 0; i < 8; ++i) { f[i] = lo[i]; f[i + 8] = hi[i]; }
  return f;
}

// stage one 64x32-half W tile into LDS (256 threads x 16B)
__device__ __forceinline__ void stage_w_tile(const _Float16* __restrict__ W,
                                             _Float16* sbuf, int n0, int K, int k0) {
  const int tid  = threadIdx.x;
  const int srow = tid >> 2;          // 0..63
  const int sch  = tid & 3;           // 16B chunk within the 64B row
  const _Float16* g = W + (size_t)(n0 + srow) * K + k0 + sch * 8;
  _Float16* d = sbuf + srow * SW_STRIDE + sch * 8;
#if GEMM_ASYNC_LDS
  __builtin_amdgcn_global_load_async_to_lds_b128(
      (AS1v4i*)(size_t)g, (AS3v4i*)(unsigned)(size_t)d, 0, 0);
#else
  *(v8h*)d = *(const v8h*)g;
#endif
}

__device__ __forceinline__ void async_wait_all() {
#if GEMM_ASYNC_LDS
#  if defined(__has_builtin) && __has_builtin(__builtin_amdgcn_s_wait_asynccnt)
  __builtin_amdgcn_s_wait_asynccnt(0);
#  else
  asm volatile("s_wait_asynccnt 0x0" ::: "memory");
#  endif
#endif
}

// ---------------------------------------------------------------- converters

__global__ void k_f32_to_f16(const float* __restrict__ in, _Float16* __restrict__ out,
                             size_t n) {
  size_t i = (size_t)blockIdx.x * 256 + threadIdx.x;
  if (i < n) out[i] = (_Float16)in[i];
}

// up_w (E,D,R) -> Uflat (D x (E*R)) row-major: Uflat[d][e*R+r] = up_w[e][d][r]
__global__ void k_up_permute(const float* __restrict__ up, _Float16* __restrict__ out) {
  int i = blockIdx.x * 256 + threadIdx.x;          // 0 .. E*D*R-1
  int r = i & (RADP - 1);
  int d = (i >> 6) & (DMODEL - 1);
  int e = i >> 16;
  out[(size_t)d * NADP + e * RADP + r] =
      (_Float16)up[(size_t)e * DMODEL * RADP + (size_t)d * RADP + r];
}

// ---------------------------------------------------------------- layernorm

__global__ __launch_bounds__(256)
void k_layernorm_h(const float* __restrict__ X, const float* __restrict__ w,
                   const float* __restrict__ b, _Float16* __restrict__ out) {
  __shared__ float red[2][8];
  const int m = blockIdx.x;
  const float* x = X + (size_t)m * DMODEL;
  float s = 0.f, s2 = 0.f;
  for (int i = threadIdx.x; i < DMODEL; i += 256) { float v = x[i]; s += v; s2 += v * v; }
#pragma unroll
  for (int o = 16; o; o >>= 1) { s += __shfl_xor(s, o, 32); s2 += __shfl_xor(s2, o, 32); }
  if ((threadIdx.x & 31) == 0) { red[0][threadIdx.x >> 5] = s; red[1][threadIdx.x >> 5] = s2; }
  __syncthreads();
  float ts = 0.f, ts2 = 0.f;
#pragma unroll
  for (int i = 0; i < 8; ++i) { ts += red[0][i]; ts2 += red[1][i]; }
  const float mu   = ts * (1.f / DMODEL);
  const float var  = ts2 * (1.f / DMODEL) - mu * mu;
  const float rstd = rsqrtf(var + 1e-5f);
  for (int i = threadIdx.x; i < DMODEL; i += 256)
    out[(size_t)m * DMODEL + i] = (_Float16)((x[i] - mu) * rstd * w[i] + b[i]);
}

// ---------------------------------------------------------------- main GEMM

enum EpMode { EP_QKV = 0, EP_RESID = 1, EP_DOWN = 2, EP_UP = 3, EP_GELU = 4, EP_FINAL = 5 };

// C(MTOK x N) = A(MTOK x K f16) @ W(N x K f16)^T
// 8 waves/block; waves share the 64-wide N range -> W tile staged in LDS (double buffer).
template <int MODE>
__global__ __launch_bounds__(256)
void k_gemm(const _Float16* __restrict__ A, const _Float16* __restrict__ W,
            int K, int N,
            const float* __restrict__ bias,
            const float* __restrict__ aux0,   // RESID: x_in | UP: gate_bias | FINAL: x1
            const float* __restrict__ aux1,   // DOWN: gates | FINAL: y
            float* __restrict__ outF,
            _Float16* __restrict__ outH,      // QKV: Q | DOWN/GELU: f16 out
            _Float16* __restrict__ outH2,     // QKV: K
            _Float16* __restrict__ outH3) {   // QKV: V^T
  __shared__ __align__(16) _Float16 sW[2][SW_ELEMS];
  const int wave = threadIdx.x >> 5;
  const int lane = threadIdx.x & 31;
  const int row0 = (blockIdx.y * 8 + wave) * 16;
  const int n0   = blockIdx.x * 64;
  const int nsteps = K / 32;

  stage_w_tile(W, sW[0], n0, K, 0);
  v8f acc[4] = {};
  for (int s = 0; s < nsteps; ++s) {
    const int cur = s & 1;
    async_wait_all();                 // own wave's staged slice has landed
    __syncthreads();                  // whole tile visible; prev reads of buf done
    if (s + 1 < nsteps) stage_w_tile(W, sW[1 - cur], n0, K, (s + 1) * 32);
    if (s + 1 < nsteps)
      __builtin_prefetch((const void*)(A + (size_t)row0 * K + (s + 1) * 32), 0, 1);
    v16h a = load_fragA(A, row0, K, s * 32);
#pragma unroll
    for (int j = 0; j < 4; ++j)
      acc[j] = wmma32(a, load_fragB_lds(sW[cur], j * 16), acc[j]);
    __syncthreads();                  // buf reads done before it is re-staged
  }

  const int hi = lane >> 4;
  const int nl = lane & 15;
#pragma unroll
  for (int j = 0; j < 4; ++j) {
#pragma unroll
    for (int r = 0; r < 8; ++r) {
      const int m = row0 + r + 8 * hi;
      const int n = n0 + j * 16 + nl;
      float v = acc[j][r];
      if constexpr (MODE == EP_QKV) {
        v += bias[n];
        const int which = n >> 10;       // uniform per 16-wide tile
        const int f = n & 1023;
        const int h = f >> 6, dd = f & 63;
        const int l = m >> 6, b = m & 63;
        const int bh = b * NHEAD + h;
        if (which == 0)      outH [((size_t)bh * LSEQ + l) * HDIM + dd] = (_Float16)v;
        else if (which == 1) outH2[((size_t)bh * LSEQ + l) * HDIM + dd] = (_Float16)v;
        else                 outH3[((size_t)bh * HDIM + dd) * LSEQ + l] = (_Float16)v;
      } else if constexpr (MODE == EP_RESID) {
        outF[(size_t)m * N + n] = v + bias[n] + aux0[(size_t)m * N + n];
      } else if constexpr (MODE == EP_DOWN) {
        v += bias[n];
        v = v > 0.f ? v : 0.f;                         // ReLU
        v *= aux1[(m & 63) * NEXP + (n >> 6)];         // fold gates[b,e] into h
        outH[(size_t)m * N + n] = (_Float16)v;
      } else if constexpr (MODE == EP_UP) {
        v = (v + aux0[(size_t)(m & 63) * N + n]) * SCALE_ADAPTER;
        outF[(size_t)m * N + n] = v;
      } else if constexpr (MODE == EP_GELU) {
        v += bias[n];
        const float s = 1.f / (1.f + __expf(-1.702f * v));   // QuickGELU
        outH[(size_t)m * N + n] = (_Float16)(v * s);
      } else {  // EP_FINAL : out = c_proj + bias + x1 + y
        outF[(size_t)m * N + n] =
            v + bias[n] + aux0[(size_t)m * N + n] + aux1[(size_t)m * N + n];
      }
    }
  }
}

// ---------------------------------------------------------------- attention

// grid (B*H, LSEQ/16), one 32-lane wave per 16-row tile of one head.
__global__ __launch_bounds__(32)
void k_attention(const _Float16* __restrict__ Qh, const _Float16* __restrict__ Kh,
                 const _Float16* __restrict__ Vth, _Float16* __restrict__ Oh) {
  __shared__ __align__(16) float    s_sc[16 * LSEQ];   // 16 KB
  __shared__ __align__(16) _Float16 s_p [16 * LSEQ];   //  8 KB
  const int bh   = blockIdx.x;
  const int rt   = blockIdx.y;
  const int lane = threadIdx.x;
  const int hi = lane >> 4, nl = lane & 15;
  const int b = bh >> 4, h = bh & 15;

  const _Float16* Q  = Qh  + (size_t)bh * LSEQ * HDIM;
  const _Float16* Km = Kh  + (size_t)bh * LSEQ * HDIM;
  const _Float16* Vt = Vth + (size_t)bh * HDIM * LSEQ;

  // scores tile-row: S = (Q_rt @ K^T) * 1/sqrt(hd)
  for (int nt = 0; nt < LSEQ / 16; ++nt) {
    v8f acc = {};
    acc = wmma32(load_fragA(Q, rt * 16, HDIM, 0),  load_fragB(Km, nt * 16, HDIM, 0),  acc);
    acc = wmma32(load_fragA(Q, rt * 16, HDIM, 32), load_fragB(Km, nt * 16, HDIM, 32), acc);
#pragma unroll
    for (int r = 0; r < 8; ++r)
      s_sc[(r + 8 * hi) * LSEQ + nt * 16 + nl] = acc[r] * 0.125f;   // 1/sqrt(64)
  }
  __syncthreads();

  // row softmax: lane (hi,nl) handles half-row nl, cols hi*128..+127
  {
    const float* row = s_sc + nl * LSEQ + hi * 128;
    float mx = -3.0e38f;
    for (int c = 0; c < 128; ++c) mx = fmaxf(mx, row[c]);
    mx = fmaxf(mx, __shfl_xor(mx, 16, 32));
    float sum = 0.f;
    for (int c = 0; c < 128; ++c) sum += __expf(row[c] - mx);
    sum += __shfl_xor(sum, 16, 32);
    const float inv = 1.f / sum;
    _Float16* prow = s_p + nl * LSEQ + hi * 128;
    for (int c = 0; c < 128; ++c) prow[c] = (_Float16)(__expf(row[c] - mx) * inv);
  }
  __syncthreads();

  // O_rt = P @ V  (V supplied transposed -> contiguous B fragments)
#pragma unroll
  for (int dt = 0; dt < HDIM / 16; ++dt) {
    v8f acc = {};
    for (int m0 = 0; m0 < LSEQ; m0 += 32)
      acc = wmma32(load_fragA(s_p, 0, LSEQ, m0), load_fragB(Vt, dt * 16, LSEQ, m0), acc);
#pragma unroll
    for (int r = 0; r < 8; ++r) {
      const int l = rt * 16 + r + 8 * hi;
      Oh[((size_t)(l * BATCH + b)) * DMODEL + h * HDIM + dt * 16 + nl] = (_Float16)acc[r];
    }
  }
}

// ---------------------------------------------------------------- gating

__global__ __launch_bounds__(64)
void k_gating(const float* __restrict__ x1, const float* __restrict__ wg,
              float* __restrict__ gates, float* __restrict__ moe_loss_out) {
  __shared__ float gsh[BATCH][NEXP];
  const int b = threadIdx.x;
  float lg[NEXP] = {};
  for (int d = 0; d < DMODEL; ++d) {
    const float xv = x1[(size_t)b * DMODEL + d];
#pragma unroll
    for (int e = 0; e < NEXP; ++e) lg[e] += xv * wg[d * NEXP + e];
  }
  int i1 = 0;
#pragma unroll
  for (int e = 1; e < NEXP; ++e) if (lg[e] > lg[i1]) i1 = e;
  int i2 = (i1 == 0) ? 1 : 0;
#pragma unroll
  for (int e = 0; e < NEXP; ++e) if (e != i1 && lg[e] > lg[i2]) i2 = e;
  const float e2 = __expf(lg[i2] - lg[i1]);
  const float g1 = 1.f / (1.f + e2);
  const float g2 = e2 / (1.f + e2);
#pragma unroll
  for (int e = 0; e < NEXP; ++e) gsh[b][e] = 0.f;
  gsh[b][i1] = g1; gsh[b][i2] = g2;
#pragma unroll
  for (int e = 0; e < NEXP; ++e) gates[b * NEXP + e] = gsh[b][e];
  __syncthreads();
  if (b == 0) {
    float imp[NEXP], ld[NEXP];
#pragma unroll
    for (int e = 0; e < NEXP; ++e) { imp[e] = 0.f; ld[e] = 0.f; }
    for (int bb = 0; bb < BATCH; ++bb)
#pragma unroll
      for (int e = 0; e < NEXP; ++e) {
        const float g = gsh[bb][e];
        imp[e] += g; ld[e] += (g > 0.f) ? 1.f : 0.f;
      }
    float loss = 0.f;
#pragma unroll
    for (int t = 0; t < 2; ++t) {
      const float* a = t ? ld : imp;
      float mu = 0.f;
      for (int e = 0; e < NEXP; ++e) mu += a[e];
      mu *= (1.f / NEXP);
      float var = 0.f;
      for (int e = 0; e < NEXP; ++e) { const float d = a[e] - mu; var += d * d; }
      var *= (1.f / (NEXP - 1));                 // ddof=1
      loss += var / (mu * mu + 1e-10f);
    }
    moe_loss_out[0] = loss * 0.01f;
  }
}

__global__ void k_gate_bias(const float* __restrict__ gates, const float* __restrict__ up_b,
                            float* __restrict__ gb) {
  const int i = blockIdx.x * 256 + threadIdx.x;    // B*DMODEL
  const int b = i >> 10, d = i & 1023;
  float s = 0.f;
#pragma unroll
  for (int e = 0; e < NEXP; ++e) s += gates[b * NEXP + e] * up_b[e * DMODEL + d];
  gb[i] = s;
}

}  // namespace

// ---------------------------------------------------------------- launcher

extern "C" void kernel_launch(void* const* d_in, const int* in_sizes, int n_in,
                              void* d_out, int out_size, void* d_ws, size_t ws_size,
                              hipStream_t stream) {
  (void)in_sizes; (void)n_in; (void)out_size; (void)ws_size;

  const float* x         = (const float*)d_in[0];
  const float* ln1_w     = (const float*)d_in[1];
  const float* ln1_b     = (const float*)d_in[2];
  const float* in_proj_w = (const float*)d_in[3];
  const float* in_proj_b = (const float*)d_in[4];
  const float* out_proj_w= (const float*)d_in[5];
  const float* out_proj_b= (const float*)d_in[6];
  const float* ln2_w     = (const float*)d_in[7];
  const float* ln2_b     = (const float*)d_in[8];
  const float* c_fc_w    = (const float*)d_in[9];
  const float* c_fc_b    = (const float*)d_in[10];
  const float* c_proj_w  = (const float*)d_in[11];
  const float* c_proj_b  = (const float*)d_in[12];
  const float* w_gate    = (const float*)d_in[13];
  const float* down_w    = (const float*)d_in[14];
  const float* down_b    = (const float*)d_in[15];
  const float* up_w      = (const float*)d_in[16];
  const float* up_b      = (const float*)d_in[17];
  float* out = (float*)d_out;

  size_t off = 0;
  auto alloc = [&](size_t bytes) {
    void* p = (char*)d_ws + off;
    off += (bytes + 255) & ~(size_t)255;
    return p;
  };
  _Float16* inW_h   = (_Float16*)alloc((size_t)3 * DMODEL * DMODEL * 2);
  _Float16* outW_h  = (_Float16*)alloc((size_t)DMODEL * DMODEL * 2);
  _Float16* cfcW_h  = (_Float16*)alloc((size_t)DFF * DMODEL * 2);
  _Float16* cprW_h  = (_Float16*)alloc((size_t)DMODEL * DFF * 2);
  _Float16* downW_h = (_Float16*)alloc((size_t)NADP * DMODEL * 2);
  _Float16* upW_h   = (_Float16*)alloc((size_t)DMODEL * NADP * 2);
  _Float16* Xh      = (_Float16*)alloc((size_t)MTOK * DMODEL * 2);   // ln1 out, reused for ln2 out
  _Float16* X1h     = (_Float16*)alloc((size_t)MTOK * DMODEL * 2);   // f16 of x1 (adapter input)
  _Float16* Qh      = (_Float16*)alloc((size_t)MTOK * DMODEL * 2);
  _Float16* Kh      = (_Float16*)alloc((size_t)MTOK * DMODEL * 2);
  _Float16* Vth     = (_Float16*)alloc((size_t)MTOK * DMODEL * 2);
  _Float16* Oh      = (_Float16*)alloc((size_t)MTOK * DMODEL * 2);
  _Float16* Hh      = (_Float16*)alloc((size_t)MTOK * NADP * 2);
  _Float16* H2h     = (_Float16*)alloc((size_t)MTOK * DFF * 2);
  float*    x1      = (float*)alloc((size_t)MTOK * DMODEL * 4);
  float*    Yf      = (float*)alloc((size_t)MTOK * DMODEL * 4);
  float*    gates   = (float*)alloc((size_t)BATCH * NEXP * 4);
  float*    gbias   = (float*)alloc((size_t)BATCH * DMODEL * 4);

  const dim3 blk256(256);
  auto cvt = [&](const float* src, _Float16* dst, size_t n) {
    k_f32_to_f16<<<dim3((unsigned)((n + 255) / 256)), blk256, 0, stream>>>(src, dst, n);
  };

  // weight conversion (identity layouts) + up_w permutation
  cvt(in_proj_w,  inW_h,   (size_t)3 * DMODEL * DMODEL);
  cvt(out_proj_w, outW_h,  (size_t)DMODEL * DMODEL);
  cvt(c_fc_w,     cfcW_h,  (size_t)DFF * DMODEL);
  cvt(c_proj_w,   cprW_h,  (size_t)DMODEL * DFF);
  cvt(down_w,     downW_h, (size_t)NADP * DMODEL);
  k_up_permute<<<dim3(NEXP * DMODEL * RADP / 256), blk256, 0, stream>>>(up_w, upW_h);

  // ln1 -> f16
  k_layernorm_h<<<dim3(MTOK), blk256, 0, stream>>>(x, ln1_w, ln1_b, Xh);

  // QKV projection (scatter to Q / K / V^T, f16)
  k_gemm<EP_QKV><<<dim3(3 * DMODEL / 64, MTOK / 128), blk256, 0, stream>>>(
      Xh, inW_h, DMODEL, 3 * DMODEL, in_proj_b, nullptr, nullptr,
      nullptr, Qh, Kh, Vth);

  // attention
  k_attention<<<dim3(BATCH * NHEAD, LSEQ / 16), dim3(32), 0, stream>>>(Qh, Kh, Vth, Oh);

  // out_proj + residual -> x1 (f32)
  k_gemm<EP_RESID><<<dim3(DMODEL / 64, MTOK / 128), blk256, 0, stream>>>(
      Oh, outW_h, DMODEL, DMODEL, out_proj_b, x, nullptr,
      x1, nullptr, nullptr, nullptr);

  // noisy-top-k gating (eval path) on token 0; moe_loss -> out[L*B*D]
  k_gating<<<dim3(1), dim3(BATCH), 0, stream>>>(x1, w_gate, gates, out + (size_t)MTOK * DMODEL);
  k_gate_bias<<<dim3(BATCH * DMODEL / 256), blk256, 0, stream>>>(gates, up_b, gbias);

  // adapter path: x1 -> f16, down (ReLU * gate), up (+gates@up_b, *0.1) -> Yf
  cvt(x1, X1h, (size_t)MTOK * DMODEL);
  k_gemm<EP_DOWN><<<dim3(NADP / 64, MTOK / 128), blk256, 0, stream>>>(
      X1h, downW_h, DMODEL, NADP, down_b, nullptr, gates,
      nullptr, Hh, nullptr, nullptr);
  k_gemm<EP_UP><<<dim3(DMODEL / 64, MTOK / 128), blk256, 0, stream>>>(
      Hh, upW_h, NADP, DMODEL, nullptr, gbias, nullptr,
      Yf, nullptr, nullptr, nullptr);

  // MLP: ln2 -> c_fc (QuickGELU) -> c_proj fused with final add: out = x1 + mlp + y
  k_layernorm_h<<<dim3(MTOK), blk256, 0, stream>>>(x1, ln2_w, ln2_b, Xh);
  k_gemm<EP_GELU><<<dim3(DFF / 64, MTOK / 128), blk256, 0, stream>>>(
      Xh, cfcW_h, DMODEL, DFF, c_fc_b, nullptr, nullptr,
      nullptr, H2h, nullptr, nullptr);
  k_gemm<EP_FINAL><<<dim3(DMODEL / 64, MTOK / 128), blk256, 0, stream>>>(
      H2h, cprW_h, DFF, DMODEL, c_proj_b, x1, Yf,
      out, nullptr, nullptr, nullptr);
}
